// LstmDecoder_21483426414791
// MI455X (gfx1250) — compile-verified, matrix operand order checked by reading, not run on hardware
//
#include <hip/hip_runtime.h>
#include <hip/hip_bf16.h>

// ---------------------------------------------------------------------------
// Problem constants (match reference)
// ---------------------------------------------------------------------------
#define VOC 16000
#define DD  1024
#define HH  1024
#define HS_ 1024
#define SS  256
#define BB  32
#define TT  128
#define NEGV (-9999.9f)
#define EPSV (1e-8f)

typedef __attribute__((ext_vector_type(16))) __bf16 v16bf;
typedef __attribute__((ext_vector_type(8)))  __bf16 v8bf;
typedef __attribute__((ext_vector_type(8)))  float  v8f;
typedef unsigned short ushort_t;

__device__ __forceinline__ ushort_t f32_to_bf16(float f) {
  unsigned int u = __float_as_uint(f);
  unsigned int r = 0x7FFFu + ((u >> 16) & 1u);   // round-to-nearest-even
  u += r;
  return (ushort_t)(u >> 16);
}

// ---------------------------------------------------------------------------
// Weight swizzle: build fragment-major bf16 B-matrix tiles (K x N logical).
// Fragment = 32x16 (KxN) tile; per ISA 7.12.2 16-bit layout assumption:
//   lane L holds column n = L%16; K offset = (L<16 ? 0 : 8);
//   element j -> k = kt*32 + off + (j<8 ? j : j+8).
// ---------------------------------------------------------------------------
__global__ void k_swizzleB(const float* __restrict__ W0, const float* __restrict__ W1,
                           int K0, int ld0, int ld1, int K, int N, int trans,
                           ushort_t* __restrict__ out) {
  size_t e = (size_t)blockIdx.x * blockDim.x + threadIdx.x;
  size_t total = (size_t)K * (size_t)N;
  if (e >= total) return;
  int j    = (int)(e & 15);
  int lane = (int)((e >> 4) & 31);
  size_t frag = e >> 9;
  int ntn = N >> 4;
  int kt = (int)(frag / ntn);
  int nt = (int)(frag % ntn);
  int n = nt * 16 + (lane & 15);
  int k = kt * 32 + ((lane < 16) ? 0 : 8) + ((j < 8) ? j : (j + 8));
  float v;
  if (trans)        v = W0[(size_t)k * N + n];
  else if (k < K0)  v = W0[(size_t)n * ld0 + k];
  else              v = W1[(size_t)n * ld1 + (k - K0)];
  out[e] = f32_to_bf16(v);
}

__global__ void k_f2bf(const float* __restrict__ x, ushort_t* __restrict__ y, size_t n) {
  size_t i = (size_t)blockIdx.x * blockDim.x + threadIdx.x;
  if (i < n) y[i] = f32_to_bf16(x[i]);
}
__global__ void k_zerof(float* __restrict__ p, int n) {
  int i = blockIdx.x * blockDim.x + threadIdx.x;
  if (i < n) p[i] = 0.f;
}
__global__ void k_zerou(ushort_t* __restrict__ p, int n) {
  int i = blockIdx.x * blockDim.x + threadIdx.x;
  if (i < n) p[i] = 0;
}

__device__ __forceinline__ v16bf pack_a(const ushort_t* Ar) {
  v8bf lo = *reinterpret_cast<const v8bf*>(Ar);
  v8bf hi = *reinterpret_cast<const v8bf*>(Ar + 16);
  v16bf a;
#pragma unroll
  for (int i = 0; i < 8; ++i) { a[i] = lo[i]; a[i + 8] = hi[i]; }
  return a;
}

// ---------------------------------------------------------------------------
// WMMA GEMM: C(MxN,f32) = A(MxK, bf16 row-major) * Bfrag(KxN, fragment-major)
// One wave handles a 32x16 output slab (two 16x16 M-tiles sharing each B
// fragment).  grid = (N/16, M/32).
// ---------------------------------------------------------------------------
__global__ __launch_bounds__(32)
void k_gemm_bf16(const ushort_t* __restrict__ A, const ushort_t* __restrict__ Bfrag,
                 float* __restrict__ C, int N, int K) {
  const int lane = threadIdx.x & 31;
  const int nt = blockIdx.x;
  const int ntn = N >> 4;
  const int r0 = blockIdx.y * 32 + (lane & 15);      // row in M-tile 0
  const int koff = (lane < 16) ? 0 : 8;
  v8f acc0 = {0.f, 0.f, 0.f, 0.f, 0.f, 0.f, 0.f, 0.f};
  v8f acc1 = {0.f, 0.f, 0.f, 0.f, 0.f, 0.f, 0.f, 0.f};
  const int ktmax = K >> 5;
  for (int kt = 0; kt < ktmax; ++kt) {
    const ushort_t* Ar0 = A + (size_t)r0 * K + kt * 32 + koff;
    v16bf a0 = pack_a(Ar0);
    v16bf a1 = pack_a(Ar0 + (size_t)16 * K);
    const v16bf b = *reinterpret_cast<const v16bf*>(
        Bfrag + (((size_t)kt * ntn + nt) * 32 + lane) * 16);
    if (kt + 8 < ktmax)
      __builtin_prefetch(Bfrag + (((size_t)(kt + 8) * ntn + nt) * 32 + lane) * 16, 0, 0);
    acc0 = __builtin_amdgcn_wmma_f32_16x16x32_bf16(false, a0, false, b,
                                                   (short)0, acc0, false, false);
    acc1 = __builtin_amdgcn_wmma_f32_16x16x32_bf16(false, a1, false, b,
                                                   (short)0, acc1, false, false);
  }
  const int n = nt * 16 + (lane & 15);
  const int mb = blockIdx.y * 32 + ((lane < 16) ? 0 : 8);
#pragma unroll
  for (int r = 0; r < 8; ++r) {
    C[(size_t)(mb + r) * N + n]      = acc0[r];
    C[(size_t)(mb + 16 + r) * N + n] = acc1[r];
  }
}

// ---------------------------------------------------------------------------
// Same GEMM, but B fragments are software-pipelined through LDS with the
// CDNA5 async copy path (global_load_async_to_lds_b128 + s_wait_asynccnt),
// 2-deep double buffer.  One wave per block -> private 2KB staging buffer.
// Used for the dominant tied-vocab logit GEMM (32MB weight stream per step).
// ---------------------------------------------------------------------------
__global__ __launch_bounds__(32)
void k_gemm_bf16_async(const ushort_t* __restrict__ A, const ushort_t* __restrict__ Bfrag,
                       float* __restrict__ C, int N, int K) {
  __shared__ __align__(32) ushort_t sbuf[2][32][16];   // 2 x 1KB B-fragment buffers
  const int lane = threadIdx.x & 31;
  const int nt = blockIdx.x;
  const int ntn = N >> 4;
  const int r0 = blockIdx.y * 32 + (lane & 15);
  const int koff = (lane < 16) ? 0 : 8;
  const int ktmax = K >> 5;
  const unsigned long long bbase = (unsigned long long)(const void*)Bfrag;
  unsigned lds[2];
  lds[0] = (unsigned)(size_t)&sbuf[0][lane][0];
  lds[1] = (unsigned)(size_t)&sbuf[1][lane][0];

  // prologue: stage kt=0 fragment (2 x b128 per lane, ASYNCcnt += 2)
  {
    unsigned goff = (unsigned)((((size_t)0 * ntn + nt) * 32 + lane) * 32);
    asm volatile("global_load_async_to_lds_b128 %0, %1, %2\n\t"
                 "global_load_async_to_lds_b128 %0, %1, %2 offset:16"
                 :: "v"(lds[0]), "v"(goff), "s"(bbase) : "memory");
  }

  v8f acc0 = {0.f, 0.f, 0.f, 0.f, 0.f, 0.f, 0.f, 0.f};
  v8f acc1 = {0.f, 0.f, 0.f, 0.f, 0.f, 0.f, 0.f, 0.f};
  for (int kt = 0; kt < ktmax; ++kt) {
    if (kt + 1 < ktmax) {
      unsigned goff = (unsigned)((((size_t)(kt + 1) * ntn + nt) * 32 + lane) * 32);
      asm volatile("global_load_async_to_lds_b128 %0, %1, %2\n\t"
                   "global_load_async_to_lds_b128 %0, %1, %2 offset:16"
                   :: "v"(lds[(kt + 1) & 1]), "v"(goff), "s"(bbase) : "memory");
      asm volatile("s_wait_asynccnt 0x2" ::: "memory");   // current buffer ready
    } else {
      asm volatile("s_wait_asynccnt 0x0" ::: "memory");
    }
    const ushort_t* Ar0 = A + (size_t)r0 * K + kt * 32 + koff;
    v16bf a0 = pack_a(Ar0);
    v16bf a1 = pack_a(Ar0 + (size_t)16 * K);
    const v16bf b = *reinterpret_cast<const v16bf*>(&sbuf[kt & 1][lane][0]);
    acc0 = __builtin_amdgcn_wmma_f32_16x16x32_bf16(false, a0, false, b,
                                                   (short)0, acc0, false, false);
    acc1 = __builtin_amdgcn_wmma_f32_16x16x32_bf16(false, a1, false, b,
                                                   (short)0, acc1, false, false);
  }
  const int n = nt * 16 + (lane & 15);
  const int mb = blockIdx.y * 32 + ((lane < 16) ? 0 : 8);
#pragma unroll
  for (int r = 0; r < 8; ++r) {
    C[(size_t)(mb + r) * N + n]      = acc0[r];
    C[(size_t)(mb + 16 + r) * N + n] = acc1[r];
  }
}

// ---------------------------------------------------------------------------
// Embedding + [emb | h0_prev] bf16 concat  (A0: 32 x 2048)
// ---------------------------------------------------------------------------
__global__ void k_embed(const float* __restrict__ E, const int* __restrict__ sot,
                        const int* __restrict__ target, const float* __restrict__ h0,
                        ushort_t* __restrict__ A0, int t) {
  int idx = blockIdx.x * blockDim.x + threadIdx.x;
  if (idx >= BB * 2 * DD) return;
  int b = idx >> 11, d = idx & 2047;
  float v;
  if (d < DD) {
    int tok = (t == 0) ? sot[0] : target[(size_t)(t - 1) * BB + b];
    v = E[(size_t)tok * DD + d];
  } else {
    v = h0[(size_t)b * HH + (d - DD)];
  }
  A0[idx] = f32_to_bf16(v);
}

// ---------------------------------------------------------------------------
// LSTM cell pointwise (gate order i,f,g,o). Writes fp32 state + bf16 copies.
// ---------------------------------------------------------------------------
__global__ void k_cell(const float* __restrict__ g, const float* __restrict__ bias,
                       float* __restrict__ h, float* __restrict__ c,
                       ushort_t* d1, int ld1, ushort_t* d2, int ld2) {
  int idx = blockIdx.x * blockDim.x + threadIdx.x;
  if (idx >= BB * HH) return;
  int b = idx >> 10, j = idx & 1023;
  const float* gr = g + (size_t)b * 4 * HH;
  float gi = gr[j]          + bias[j];
  float gf = gr[HH + j]     + bias[HH + j];
  float gg = gr[2 * HH + j] + bias[2 * HH + j];
  float go = gr[3 * HH + j] + bias[3 * HH + j];
  float si = 1.f / (1.f + expf(-gi));
  float sf = 1.f / (1.f + expf(-gf));
  float so = 1.f / (1.f + expf(-go));
  float cn = sf * c[idx] + si * tanhf(gg);
  float hn = so * tanhf(cn);
  c[idx] = cn;
  h[idx] = hn;
  ushort_t hb = f32_to_bf16(hn);
  if (d1) d1[(size_t)b * ld1 + j] = hb;
  if (d2) d2[(size_t)b * ld2 + j] = hb;
}

// ---------------------------------------------------------------------------
// Attention scores: one wave per (s,b); lanes partition H, shfl reduction.
// ---------------------------------------------------------------------------
__global__ void k_scores(const float* __restrict__ Whs, const float* __restrict__ h1,
                         float* __restrict__ sc) {
  int w = blockIdx.x * (blockDim.x >> 5) + (threadIdx.x >> 5);
  int lane = threadIdx.x & 31;
  if (w >= SS * BB) return;
  int s = w >> 5, b = w & 31;   // BB == 32
  const float* wp = Whs + ((size_t)s * BB + b) * HH;
  const float* hp = h1 + (size_t)b * HH;
  float acc = 0.f;
  for (int hh = lane; hh < HH; hh += 32) acc += wp[hh] * hp[hh];
#pragma unroll
  for (int off = 16; off > 0; off >>= 1) acc += __shfl_down(acc, off, 32);
  if (lane == 0) sc[(size_t)s * BB + b] = acc;
}

// ---------------------------------------------------------------------------
// Masked softmax over S per batch element. Writes almt (S,B) + out (B,S).
// ---------------------------------------------------------------------------
__global__ void k_attn_softmax(const float* __restrict__ sc,
                               const unsigned char* __restrict__ mask,
                               float* __restrict__ almt, float* __restrict__ out_al) {
  __shared__ float sh[SS];
  int b = blockIdx.x;
  int s = threadIdx.x;            // blockDim == SS == 256
  float v = mask[(size_t)b * SS + s] ? sc[(size_t)s * BB + b] : NEGV;
  sh[s] = v; __syncthreads();
  for (int k = SS / 2; k > 0; k >>= 1) {
    if (s < k) sh[s] = fmaxf(sh[s], sh[s + k]);
    __syncthreads();
  }
  float mx = sh[0]; __syncthreads();
  float e = expf(v - mx);
  sh[s] = e; __syncthreads();
  for (int k = SS / 2; k > 0; k >>= 1) {
    if (s < k) sh[s] += sh[s + k];
    __syncthreads();
  }
  float a = e / sh[0];
  almt[(size_t)s * BB + b] = a;
  out_al[(size_t)b * SS + s] = a;
}

// ---------------------------------------------------------------------------
// Context sums: ctx_emb (f32) and ctx (bf16 into Acat tail). thread = (b,d).
// ---------------------------------------------------------------------------
__global__ void k_ctx(const float* __restrict__ almt, const float* __restrict__ src_out,
                      const float* __restrict__ src_emb, float* __restrict__ ctxe,
                      ushort_t* __restrict__ Acat) {
  int idx = blockIdx.x * blockDim.x + threadIdx.x;
  if (idx >= BB * HS_) return;
  int b = idx >> 10, d = idx & 1023;
  float a1 = 0.f, a2 = 0.f;
  for (int s = 0; s < SS; ++s) {
    float w = almt[(size_t)s * BB + b];
    size_t base = ((size_t)s * BB + b) * HS_ + d;
    a1 += w * src_out[base];
    a2 += w * src_emb[base];
  }
  ctxe[idx] = a2;
  Acat[(size_t)b * 2 * HH + HH + d] = f32_to_bf16(a1);  // ctx -> Acat[:,1024:]
}

// ---------------------------------------------------------------------------
// Norm-controlled residual; writes logit-A operand in bf16.
// ---------------------------------------------------------------------------
__global__ void k_residual(const float* __restrict__ ctxe, const float* __restrict__ hidcat,
                           const float* __restrict__ bias, ushort_t* __restrict__ Ares) {
  __shared__ float sh[256];
  int b = blockIdx.x, tid = threadIdx.x;
  float s1 = 0.f, s2 = 0.f;
  for (int j = tid; j < HH; j += 256) {
    float ce = ctxe[(size_t)b * HH + j];
    float hv = hidcat[(size_t)b * HH + j] + bias[j];
    s1 += ce * ce;
    s2 += hv * hv;
  }
  sh[tid] = s1; __syncthreads();
  for (int k = 128; k > 0; k >>= 1) { if (tid < k) sh[tid] += sh[tid + k]; __syncthreads(); }
  float n1 = sqrtf(sh[0]); __syncthreads();
  sh[tid] = s2; __syncthreads();
  for (int k = 128; k > 0; k >>= 1) { if (tid < k) sh[tid] += sh[tid + k]; __syncthreads(); }
  float nh = sqrtf(sh[0]); __syncthreads();
  float sa = n1 / (n1 + EPSV);
  float sr = n1 * 0.2f / (nh + EPSV);
  for (int j = tid; j < HH; j += 256) {
    float ce = ctxe[(size_t)b * HH + j];
    float hv = hidcat[(size_t)b * HH + j] + bias[j];
    Ares[(size_t)b * HH + j] = f32_to_bf16(ce * sa + hv * sr);
  }
}

// ---------------------------------------------------------------------------
// log_softmax over V per batch element.
// ---------------------------------------------------------------------------
__global__ void k_logsoftmax(const float* __restrict__ logits, float* __restrict__ out) {
  __shared__ float sh[256];
  int b = blockIdx.x, tid = threadIdx.x;
  const float* lr = logits + (size_t)b * VOC;
  float mx = -3.4e38f;
  for (int v = tid; v < VOC; v += 256) mx = fmaxf(mx, lr[v]);
  sh[tid] = mx; __syncthreads();
  for (int k = 128; k > 0; k >>= 1) { if (tid < k) sh[tid] = fmaxf(sh[tid], sh[tid + k]); __syncthreads(); }
  mx = sh[0]; __syncthreads();
  float sm = 0.f;
  for (int v = tid; v < VOC; v += 256) sm += expf(lr[v] - mx);
  sh[tid] = sm; __syncthreads();
  for (int k = 128; k > 0; k >>= 1) { if (tid < k) sh[tid] += sh[tid + k]; __syncthreads(); }
  float ls = logf(sh[0]); __syncthreads();
  float* orow = out + (size_t)b * VOC;
  for (int v = tid; v < VOC; v += 256) orow[v] = lr[v] - mx - ls;
}

// ---------------------------------------------------------------------------
// Host orchestration
// ---------------------------------------------------------------------------
extern "C" void kernel_launch(void* const* d_in, const int* in_sizes, int n_in,
                              void* d_out, int out_size, void* d_ws, size_t ws_size,
                              hipStream_t stream) {
  (void)in_sizes; (void)n_in; (void)out_size; (void)ws_size;
  const int*   sot      = (const int*)d_in[0];
  const float* src_emb  = (const float*)d_in[1];
  const float* src_out  = (const float*)d_in[2];
  const unsigned char* mask = (const unsigned char*)d_in[3];
  const int*   target   = (const int*)d_in[4];
  const float* E        = (const float*)d_in[5];
  const float* Wih0     = (const float*)d_in[6];
  const float* Whh0     = (const float*)d_in[7];
  const float* b0       = (const float*)d_in[8];
  const float* Wih1     = (const float*)d_in[9];
  const float* Whh1     = (const float*)d_in[10];
  const float* b1       = (const float*)d_in[11];
  const float* Wa       = (const float*)d_in[12];
  const float* Whid     = (const float*)d_in[13];
  const float* bhid     = (const float*)d_in[14];

  float* out_lp   = (float*)d_out;
  float* out_almt = out_lp + (size_t)TT * BB * VOC;

  // ---- workspace carve-up (256B aligned) ----
  char* ws = (char*)d_ws;
  size_t off = 0;
  auto carve = [&](size_t bytes) -> void* {
    off = (off + 255) & ~(size_t)255;
    void* p = ws + off;
    off += bytes;
    return p;
  };
  ushort_t* WF0   = (ushort_t*)carve((size_t)2048 * 4096 * 2);  // [Wih0;Whh0] frags
  ushort_t* WF1   = (ushort_t*)carve((size_t)2048 * 4096 * 2);
  ushort_t* WHIDF = (ushort_t*)carve((size_t)2048 * 1024 * 2);
  ushort_t* EF    = (ushort_t*)carve((size_t)1024 * VOC * 2);
  ushort_t* WAF   = (ushort_t*)carve((size_t)1024 * 1024 * 2);
  ushort_t* SOBF  = (ushort_t*)carve((size_t)SS * BB * HS_ * 2);
  float*    WHS   = (float*)carve((size_t)SS * BB * HH * 4);
  ushort_t* A0    = (ushort_t*)carve((size_t)BB * 2048 * 2);
  ushort_t* A1    = (ushort_t*)carve((size_t)BB * 2048 * 2);
  ushort_t* ACAT  = (ushort_t*)carve((size_t)BB * 2048 * 2);
  ushort_t* ARES  = (ushort_t*)carve((size_t)BB * 1024 * 2);
  float*    G0    = (float*)carve((size_t)BB * 4096 * 4);
  float*    G1    = (float*)carve((size_t)BB * 4096 * 4);
  float*    H0f   = (float*)carve((size_t)BB * HH * 4);
  float*    C0f   = (float*)carve((size_t)BB * HH * 4);
  float*    H1f   = (float*)carve((size_t)BB * HH * 4);
  float*    C1f   = (float*)carve((size_t)BB * HH * 4);
  float*    SC    = (float*)carve((size_t)SS * BB * 4);
  float*    ALM   = (float*)carve((size_t)SS * BB * 4);
  float*    CTXE  = (float*)carve((size_t)BB * HS_ * 4);
  float*    HIDC  = (float*)carve((size_t)BB * HH * 4);
  float*    LOGI  = (float*)carve((size_t)BB * VOC * 4);

  const int TPB = 256;
  auto gridFor = [](size_t n, int tpb) { return (unsigned)((n + tpb - 1) / tpb); };

  // ---- one-time (per launch) weight prep ----
  k_swizzleB<<<gridFor((size_t)2048 * 4096, TPB), TPB, 0, stream>>>(
      Wih0, Whh0, 1024, 1024, 1024, 2048, 4096, 0, WF0);
  k_swizzleB<<<gridFor((size_t)2048 * 4096, TPB), TPB, 0, stream>>>(
      Wih1, Whh1, 1024, 1024, 1024, 2048, 4096, 0, WF1);
  k_swizzleB<<<gridFor((size_t)2048 * 1024, TPB), TPB, 0, stream>>>(
      Whid, Whid, 2048, 2048, 2048, 2048, 1024, 0, WHIDF);
  k_swizzleB<<<gridFor((size_t)1024 * VOC, TPB), TPB, 0, stream>>>(
      E, E, 1024, 1024, 1024, 1024, VOC, 0, EF);
  k_swizzleB<<<gridFor((size_t)1024 * 1024, TPB), TPB, 0, stream>>>(
      Wa, Wa, 1024, 1024, 1024, 1024, 1024, 1, WAF);
  k_f2bf<<<gridFor((size_t)SS * BB * HS_, TPB), TPB, 0, stream>>>(
      src_out, SOBF, (size_t)SS * BB * HS_);

  // zero recurrent state + A1 (h1_prev half) for determinism
  k_zerof<<<gridFor(BB * HH, TPB), TPB, 0, stream>>>(H0f, BB * HH);
  k_zerof<<<gridFor(BB * HH, TPB), TPB, 0, stream>>>(C0f, BB * HH);
  k_zerof<<<gridFor(BB * HH, TPB), TPB, 0, stream>>>(H1f, BB * HH);
  k_zerof<<<gridFor(BB * HH, TPB), TPB, 0, stream>>>(C1f, BB * HH);
  k_zerou<<<gridFor(BB * 2048, TPB), TPB, 0, stream>>>(A1, BB * 2048);

  // Wh_s = src_outputs @ Wa : (8192 x 1024) * (1024 x 1024)
  k_gemm_bf16<<<dim3(1024 / 16, (SS * BB) / 32), 32, 0, stream>>>(SOBF, WAF, WHS, 1024, 1024);

  // ---- sequential decode ----
  for (int t = 0; t < TT; ++t) {
    k_embed<<<gridFor(BB * 2048, TPB), TPB, 0, stream>>>(E, sot, target, H0f, A0, t);
    k_gemm_bf16<<<dim3(4096 / 16, 1), 32, 0, stream>>>(A0, WF0, G0, 4096, 2048);
    k_cell<<<gridFor(BB * HH, TPB), TPB, 0, stream>>>(G0, b0, H0f, C0f,
                                                      A1, 2048, nullptr, 0);
    k_gemm_bf16<<<dim3(4096 / 16, 1), 32, 0, stream>>>(A1, WF1, G1, 4096, 2048);
    k_cell<<<gridFor(BB * HH, TPB), TPB, 0, stream>>>(G1, b1, H1f, C1f,
                                                      ACAT, 2048, A1 + 1024, 2048);
    k_scores<<<gridFor((size_t)SS * BB * 32, TPB), TPB, 0, stream>>>(WHS, H1f, SC);
    k_attn_softmax<<<BB, SS, 0, stream>>>(SC, mask, ALM,
                                          out_almt + (size_t)t * BB * SS);
    k_ctx<<<gridFor(BB * HS_, TPB), TPB, 0, stream>>>(ALM, src_out, src_emb, CTXE, ACAT);
    k_gemm_bf16<<<dim3(1024 / 16, 1), 32, 0, stream>>>(ACAT, WHIDF, HIDC, 1024, 2048);
    k_residual<<<BB, 256, 0, stream>>>(CTXE, HIDC, bhid, ARES);
    k_gemm_bf16_async<<<dim3(VOC / 16, 1), 32, 0, stream>>>(ARES, EF, LOGI, VOC, 1024);
    k_logsoftmax<<<BB, 256, 0, stream>>>(LOGI, out_lp + (size_t)t * BB * VOC);
  }
}
